// Multi_Head_Attention_28827820490998
// MI455X (gfx1250) — matrix operand order repaired and checked by reading, not compiled, not run on hardware
//
#include <hip/hip_runtime.h>
#include <math.h>

// ---------------------------------------------------------------------------
// MI455X (gfx1250) multi-head attention with FFT heads.
// bf16 WMMA (v_wmma_f32_16x16x32_bf16) for all GEMMs, fp32 butterflies for
// the 4x2 fft2/ifft2 (twiddles {+-1,+-i} -> adds only), fp32 softmax.
// BT=2 batches per workgroup, 8 wave32 waves, ~284KB dynamic LDS.
// Head loop split into lo(0..2)/hi(3..7) passes so each WMMA accumulator
// destination is unconditional (no PHI copies of the 128 acc VGPRs).
// ---------------------------------------------------------------------------

#define SEQ 32
#define DM  256
#define NH  8
#define FFTH 3
#define BT  2
#define THREADS 256
#define NWAVE 8

typedef __bf16 bf16_t;
typedef bf16_t v16bf __attribute__((ext_vector_type(16)));
typedef bf16_t v8bf  __attribute__((ext_vector_type(8)));
typedef float  v8f   __attribute__((ext_vector_type(8)));
typedef float  v4f   __attribute__((ext_vector_type(4)));

// ---- LDS layout (bytes). Row pitches 16B-aligned except S (33 f32). --------------------
#define PX 264                 // bf16 row pitch for X/XF/Q/K/Wt (256 + 8 pad)
#define X_OFF    0             // [64][PX] bf16   = 33792 B
#define XF_OFF   33792         // [64][PX] bf16
#define Q_OFF    67584         // [64][PX] bf16
#define K_OFF    101376        // [64][PX] bf16
#define VT_OFF   135168        // [BT][256][40] bf16 = 40960 B  (V transposed)
#define W_OFF    176128        // [3][64][PX] bf16   = 101376 B (W slab, transposed)
#define S_OFF    277504        // [BT][32][33] f32   = 8448 B   (scores)
#define A_OFF    285952        // [BT][32][40] bf16  = 5120 B   (probs)
#define SMEM_BYTES 291072
#define PD 264                 // f32 pitch for end-phase accumulator dumps

// fft2 / ifft2 over a 4x2 complex grid packed as 16 interleaved floats:
// z[c][d2] = v[4c+2d2] + i*v[4c+2d2+1].  Output packing (reference
// concat([Re,Im],-1)):  o[4k1+k2]=Re Y[k1][k2], o[4k1+2+k2]=Im Y[k1][k2].
__device__ __forceinline__ void fft4x2(const float v[16], float o[16], int inverse) {
  float u0r[4], u0i[4], u1r[4], u1i[4];
#pragma unroll
  for (int c = 0; c < 4; ++c) {
    float ar = v[4*c+0], ai = v[4*c+1], br = v[4*c+2], bi = v[4*c+3];
    u0r[c] = ar + br; u0i[c] = ai + bi;   // k2 = 0
    u1r[c] = ar - br; u1i[c] = ai - bi;   // k2 = 1
  }
  const float scale = inverse ? 0.125f : 1.0f;
#pragma unroll
  for (int k2 = 0; k2 < 2; ++k2) {
    const float* Ur = k2 ? u1r : u0r;
    const float* Ui = k2 ? u1i : u0i;
    float er = Ur[0] + Ur[2], ei = Ui[0] + Ui[2];
    float fr = Ur[0] - Ur[2], fi = Ui[0] - Ui[2];
    float gr = Ur[1] + Ur[3], gi = Ui[1] + Ui[3];
    float hr = Ur[1] - Ur[3], hi = Ui[1] - Ui[3];
    float y0r = er + gr, y0i = ei + gi;
    float y2r = er - gr, y2i = ei - gi;
    float y1r, y1i, y3r, y3i;
    if (!inverse) { y1r = fr + hi; y1i = fi - hr; y3r = fr - hi; y3i = fi + hr; }
    else          { y1r = fr - hi; y1i = fi + hr; y3r = fr + hi; y3i = fi - hr; }
    o[0*4 + k2] = y0r * scale; o[0*4 + 2 + k2] = y0i * scale;
    o[1*4 + k2] = y1r * scale; o[1*4 + 2 + k2] = y1i * scale;
    o[2*4 + k2] = y2r * scale; o[2*4 + 2 + k2] = y2i * scale;
    o[3*4 + k2] = y3r * scale; o[3*4 + 2 + k2] = y3i * scale;
  }
}

// two 8-element chunks at +0 / +16 elements -> fragment (two ds_load_b128)
__device__ __forceinline__ v16bf ldfrag1(const bf16_t* p) {
  v8bf lo = *(const v8bf*)p;
  v8bf hi = *(const v8bf*)(p + 16);
  return __builtin_shufflevector(lo, hi, 0,1,2,3,4,5,6,7,8,9,10,11,12,13,14,15);
}
// two 8-element chunks at +0 / +16 elements (A-matrix 16x32 per-lane pattern)
__device__ __forceinline__ v16bf ldfrag2(const bf16_t* p) {
  v8bf lo = *(const v8bf*)p;
  v8bf hi = *(const v8bf*)(p + 16);
  return __builtin_shufflevector(lo, hi, 0,1,2,3,4,5,6,7,8,9,10,11,12,13,14,15);
}
__device__ __forceinline__ v8f wmma_bf16(v16bf a, v16bf b, v8f c) {
  return __builtin_amdgcn_wmma_f32_16x16x32_bf16(false, a, false, b, (short)0, c,
                                                 false, false);
}

__global__ __launch_bounds__(THREADS)
void mha_fft_wmma_kernel(const float* __restrict__ xg,
                         const float* __restrict__ wq,
                         const float* __restrict__ wk,
                         const float* __restrict__ wv,
                         float* __restrict__ outg) {
  extern __shared__ __align__(16) char smem[];
  bf16_t* Xs  = (bf16_t*)(smem + X_OFF);
  bf16_t* XFs = (bf16_t*)(smem + XF_OFF);
  bf16_t* Qs  = (bf16_t*)(smem + Q_OFF);
  bf16_t* Ks  = (bf16_t*)(smem + K_OFF);
  bf16_t* VTs = (bf16_t*)(smem + VT_OFF);
  bf16_t* Wts = (bf16_t*)(smem + W_OFF);
  float*  Ss  = (float*)(smem + S_OFF);
  bf16_t* As  = (bf16_t*)(smem + A_OFF);
  float*  HiD = (float*)(smem + W_OFF);   // reused after head loop
  float*  LoD = (float*)(smem + Q_OFF);   // reused after head loop (Q+K areas)

  const int tid  = threadIdx.x;
  const int lane = tid & 31;
  const int wave = tid >> 5;
  const int b0   = blockIdx.x * BT;

  const int asel = (lane >> 4) << 3;      // A-frag: K chunk base 0 / 8
  const int bsel = (lane >> 4) << 3;      // B-frag: K chunk base 0 / 8
  const int l15  = lane & 15;
  const int msel = (lane >> 4) << 3;      // C-frag: row base 0 / 8

  // ---- Phase 0: load x, build bf16 X and bf16 XF (forward butterflies) ----
#pragma unroll
  for (int i = 0; i < 4; ++i) {
    int c  = tid + THREADS * i;          // chunk id 0..1023
    int g  = c >> 4;                     // global row 0..63 (BT*SEQ)
    int ch = c & 15;                     // 16-float chunk in the row
    const float* src = xg + (size_t)(b0 + (g >> 5)) * SEQ * DM + (g & 31) * DM + ch * 16;
    float v[16], o[16];
    const v4f* s4 = (const v4f*)src;
#pragma unroll
    for (int q = 0; q < 4; ++q) {
      v4f t = s4[q];
#pragma unroll
      for (int e = 0; e < 4; ++e) v[q * 4 + e] = t[e];
    }
    fft4x2(v, o, 0);
#pragma unroll
    for (int j = 0; j < 16; ++j) {
      Xs [g * PX + ch * 16 + j] = (bf16_t)v[j];
      XFs[g * PX + ch * 16 + j] = (bf16_t)o[j];
    }
  }
  __syncthreads();

  const v8f vzero = {0.f, 0.f, 0.f, 0.f, 0.f, 0.f, 0.f, 0.f};
  v8f accL[8], accH[8];
#pragma unroll
  for (int j = 0; j < 8; ++j) { accL[j] = vzero; accH[j] = vzero; }

  // One full head: QKV projection -> scores -> masked softmax -> O += A*V.
  // Inlined twice (lo pass / hi pass) so acc destinations are unconditional.
  auto do_head = [&](int h, const bf16_t* Xsrc, v8f* acc) {
    // ---- QKV projections over 4 slabs of 64 output columns ----
    for (int slab = 0; slab < 4; ++slab) {
      const int n0 = slab * 64;
      // stage Wq/Wk/Wv[h][:, n0:n0+64] transposed into LDS as bf16
#pragma unroll 4
      for (int i = 0; i < 48; ++i) {
        int idx4 = tid + THREADS * i;            // 0..12287 float4s
        int mat  = idx4 >> 12;                   // 0..2
        int rem  = idx4 & 4095;
        int k    = rem >> 4;                     // 0..255
        int n4   = (rem & 15) << 2;              // 0..60
        const float* wbase = (mat == 0) ? wq : ((mat == 1) ? wk : wv);
        v4f wv4 = *(const v4f*)(wbase + (size_t)h * DM * DM + (size_t)k * DM + n0 + n4);
        bf16_t* dst = Wts + mat * (64 * PX) + n4 * PX + k;
#pragma unroll
        for (int e = 0; e < 4; ++e) dst[e * PX] = (bf16_t)wv4[e];
      }
      __syncthreads();

      // prefetch a slice of the next slab while WMMAs run (global_prefetch)
      if (slab < 3) {
#pragma unroll
        for (int pi = 0; pi < 3; ++pi) {
          int idx4 = tid + THREADS * (pi * 16);
          int mat  = idx4 >> 12;
          int rem  = idx4 & 4095;
          int k    = rem >> 4;
          int n4   = (rem & 15) << 2;
          const float* wbase = (mat == 0) ? wq : ((mat == 1) ? wk : wv);
          __builtin_prefetch(wbase + (size_t)h * DM * DM + (size_t)k * DM + (n0 + 64) + n4, 0, 1);
        }
      }

      // 48 output tiles (3 mats x 4 m-tiles x 4 n-tiles); 6 per wave.
      // mat is compile-time: tiles are dealt as wave + 8*j.
#pragma unroll
      for (int j = 0; j < 6; ++j) {
        const int mat = j >> 1;                  // 0,0,1,1,2,2
        const int rem = wave + ((j & 1) << 3);   // 0..15
        const int mt  = rem >> 2, nt = rem & 3;
        v8f a8 = vzero;
        const bf16_t* arow = Xsrc + (mt * 16 + l15) * PX + asel;
        const bf16_t* brow = Wts + mat * (64 * PX) + (nt * 16 + l15) * PX + bsel;
#pragma unroll
        for (int kb = 0; kb < 8; ++kb) {
          v16bf a = ldfrag2(arow + kb * 32);
          v16bf b = ldfrag1(brow + kb * 32);
          a8 = wmma_bf16(a, b, a8);
        }
        const int ncol  = n0 + nt * 16 + l15;
        const int mrow0 = mt * 16 + msel;
        if (mat == 0) {
#pragma unroll
          for (int q = 0; q < 8; ++q) Qs[(mrow0 + q) * PX + ncol] = (bf16_t)a8[q];
        } else if (mat == 1) {
#pragma unroll
          for (int q = 0; q < 8; ++q) Ks[(mrow0 + q) * PX + ncol] = (bf16_t)a8[q];
        } else {
          const int bb = mrow0 >> 5, ml = mrow0 & 31;    // V stored transposed
          v8bf pk;
#pragma unroll
          for (int q = 0; q < 8; ++q) pk[q] = (bf16_t)a8[q];
          *(v8bf*)(VTs + bb * (256 * 40) + ncol * 40 + ml) = pk;
        }
      }
      __syncthreads();
    }

    // ---- scores S = Q*K^T / 16 : 8 (b,st,tt) tiles, one per wave ----
    {
      const int bb = wave >> 2, st = (wave >> 1) & 1, tt = wave & 1;
      v8f a8 = vzero;
      const bf16_t* arow = Qs + (bb * 32 + st * 16 + l15) * PX + asel;
      const bf16_t* brow = Ks + (bb * 32 + tt * 16 + l15) * PX + bsel;
#pragma unroll
      for (int kb = 0; kb < 8; ++kb) {
        v16bf a = ldfrag2(arow + kb * 32);
        v16bf b = ldfrag1(brow + kb * 32);     // row-major K == K^T as B-frag
        a8 = wmma_bf16(a, b, a8);
      }
      const int col = tt * 16 + l15;
      const int m0  = st * 16 + msel;
#pragma unroll
      for (int q = 0; q < 8; ++q)
        Ss[bb * 1056 + (m0 + q) * 33 + col] = a8[q] * 0.0625f;
    }
    __syncthreads();

    // ---- masked softmax over rows (atrous mask: keep (i+j) even) ----
    if (tid < BT * SEQ) {
      const int bb = tid >> 5, r = tid & 31;
      const float* row = Ss + bb * 1056 + r * 33;
      float mx = -3.0e38f;
#pragma unroll
      for (int j2 = 0; j2 < 32; ++j2)
        if (((r + j2) & 1) == 0) mx = fmaxf(mx, row[j2]);
      float e[32]; float sum = 0.f;
#pragma unroll
      for (int j2 = 0; j2 < 32; ++j2) {
        float t = (((r + j2) & 1) == 0) ? __expf(row[j2] - mx) : 0.f;
        e[j2] = t; sum += t;
      }
      const float inv = 1.0f / sum;
#pragma unroll
      for (int j2 = 0; j2 < 32; ++j2)
        As[bb * 1280 + r * 40 + j2] = (bf16_t)(e[j2] * inv);
    }
    __syncthreads();

    // ---- O += A*V : 64 tiles, 8 per wave, K=32 -> one WMMA each ----
#pragma unroll
    for (int j = 0; j < 8; ++j) {
      const int bb = j >> 2;                     // 0,0,0,0,1,1,1,1
      const int mt = (j >> 1) & 1;               // 0,0,1,1,...
      const int nt = wave + ((j & 1) << 3);      // 0..15
      v16bf a = ldfrag2(As + bb * 1280 + (mt * 16 + l15) * 40 + asel);
      v16bf b = ldfrag1(VTs + bb * (256 * 40) + (nt * 16 + l15) * 40 + bsel);
      acc[j] = wmma_bf16(a, b, acc[j]);
    }
    __syncthreads();   // before next head rewrites Wt/Q/K/VT/S/A
  };

  for (int h = 0; h < FFTH; ++h) do_head(h, Xs, accL);    // lo heads on x
  for (int h = FFTH; h < NH; ++h) do_head(h, XFs, accH);  // hi heads on fft(x)

  // ---- dump accumulators to LDS (reuse W area for hi, Q+K area for lo) ----
#pragma unroll
  for (int j = 0; j < 8; ++j) {
    const int bb = j >> 2;
    const int mt = (j >> 1) & 1;
    const int nt = wave + ((j & 1) << 3);
    const int g0  = bb * 32 + mt * 16 + msel;
    const int col = nt * 16 + l15;
#pragma unroll
    for (int q = 0; q < 8; ++q) {
      LoD[(g0 + q) * PD + col] = accL[j][q];
      HiD[(g0 + q) * PD + col] = accH[j][q];
    }
  }
  __syncthreads();

  // ---- final: out = (lo + ifft(hi)) / NUM_HEADS ----
#pragma unroll
  for (int i = 0; i < 4; ++i) {
    int c  = tid + THREADS * i;
    int g  = c >> 4;
    int ch = c & 15;
    float v[16], o[16];
    const v4f* h4 = (const v4f*)(HiD + g * PD + ch * 16);
#pragma unroll
    for (int q = 0; q < 4; ++q) {
      v4f t = h4[q];
#pragma unroll
      for (int e = 0; e < 4; ++e) v[q * 4 + e] = t[e];
    }
    fft4x2(v, o, 1);                     // includes 1/8 ifft normalization
    const float* lo = LoD + g * PD + ch * 16;
    float* dst = outg + (size_t)(b0 + (g >> 5)) * SEQ * DM + (g & 31) * DM + ch * 16;
#pragma unroll
    for (int q = 0; q < 4; ++q) {
      v4f r;
#pragma unroll
      for (int e = 0; e < 4; ++e) r[e] = (lo[q * 4 + e] + o[q * 4 + e]) * 0.125f;
      *(v4f*)(dst + q * 4) = r;
    }
  }
}

extern "C" void kernel_launch(void* const* d_in, const int* in_sizes, int n_in,
                              void* d_out, int out_size, void* d_ws, size_t ws_size,
                              hipStream_t stream) {
  const float* x  = (const float*)d_in[0];
  const float* wq = (const float*)d_in[1];
  const float* wk = (const float*)d_in[2];
  const float* wv = (const float*)d_in[3];
  float* out = (float*)d_out;

  const int B = in_sizes[0] / (SEQ * DM);
  const int nblk = B / BT;

  (void)hipFuncSetAttribute((const void*)mha_fft_wmma_kernel,
                            hipFuncAttributeMaxDynamicSharedMemorySize,
                            SMEM_BYTES);
  mha_fft_wmma_kernel<<<nblk, THREADS, SMEM_BYTES, stream>>>(x, wq, wk, wv, out);
}
